// VectorNetGlobalGraph_51582557225721
// MI455X (gfx1250) — matrix operand order repaired and checked by reading, not compiled, not run on hardware
//
#include <hip/hip_runtime.h>
#include <hip/hip_bf16.h>

// MI455X / gfx1250 fused attention:
//   k0: weight transpose + bf16 convert
//   k1: QKV projection GEMM via v_wmma_f32_16x16x32_bf16 (Q pre-scaled by 1/sqrt(H), V stored transposed)
//   k2: flash-attention (streaming online softmax), WMMA for both Q@K^T and P@V
// Workspace: 3*128*128 (wT) + 3*256*512*128 (Q,K,Vt) bf16 elements ~= 96.2 MB.

#define BATCH 256
#define SEQ   512
#define DIM   128   // D == H == 128
#define NEGV  (-1.0e12f)

typedef __attribute__((ext_vector_type(16))) __bf16 v16bf;
typedef __attribute__((ext_vector_type(8)))  float  v8f;

union ABfrag { v16bf v; unsigned int w[8]; unsigned short h[16]; };

// Native bf16 convert (lets the compiler emit v_cvt_pk_bf16_f32 instead of
// a manual RNE bit-twiddle sequence).
__device__ __forceinline__ unsigned short f2bf(float f) {
  union { __bf16 b; unsigned short u; } c;
  c.b = (__bf16)f;
  return c.u;
}
__device__ __forceinline__ unsigned pack2bf(float a, float b) {
  return (unsigned)f2bf(a) | ((unsigned)f2bf(b) << 16);
}
// A-fragment K offset for 16-bit 16x32: VGPR j holds K pair base
__device__ __forceinline__ int a_koff(int j, int hf) {
  return ((j < 4) ? (2 * j) : (16 + 2 * (j - 4))) + 8 * hf;
}

// ---------------- kernel 0: weights -> bf16, transposed wT[n][k] ----------------
__global__ void prep_weights(const float* __restrict__ wq, const float* __restrict__ wk,
                             const float* __restrict__ wv, unsigned short* __restrict__ wT) {
  int idx = blockIdx.x * blockDim.x + threadIdx.x;   // 3*128*128 threads
  int mat = idx >> 14;
  int r   = idx & 16383;
  int k = r >> 7, n = r & 127;
  const float* w = (mat == 0) ? wq : (mat == 1) ? wk : wv;
  wT[(mat << 14) + (n << 7) + k] = f2bf(w[(k << 7) + n]);
}

// ---------------- kernel 1: QKV projection ----------------
// wave -> one 16-row M tile of one matrix (Q/K/V); loops 8 column tiles, 4 k-steps each.
__global__ __launch_bounds__(256) void qkv_proj(
    const float* __restrict__ x, const unsigned short* __restrict__ wT,
    const float* __restrict__ bq, const float* __restrict__ bk, const float* __restrict__ bv,
    unsigned short* __restrict__ Qb, unsigned short* __restrict__ Kb,
    unsigned short* __restrict__ Vt) {
  const int lane  = threadIdx.x & 31;
  const int wave  = (blockIdx.x << 3) + (threadIdx.x >> 5);
  const int mat   = wave >> 13;          // / 8192 M-tiles
  const int mtile = wave & 8191;
  const int m0    = mtile << 4;
  const int mA = lane & 15, hf = lane >> 4;

  // A fragments: x rows, fp32 -> bf16 on the fly (pairs are contiguous -> float2 loads)
  ABfrag a[4];
  const float* xrow = x + (size_t)(m0 + mA) * DIM;
  #pragma unroll
  for (int ks = 0; ks < 4; ++ks)
    #pragma unroll
    for (int j = 0; j < 8; ++j) {
      int k0 = ks * 32 + a_koff(j, hf);
      float2 xv = *(const float2*)(xrow + k0);
      a[ks].w[j] = pack2bf(xv.x, xv.y);
    }

  const unsigned short* wTm = wT + (mat << 14);
  const float* bias = (mat == 0) ? bq : (mat == 1) ? bk : bv;
  const float scale = (mat == 0) ? 0.088388347648318447f : 1.0f;  // 1/sqrt(128) folded into Q
  const int b    = m0 >> 9;
  const int key0 = m0 & 511;

  for (int nt = 0; nt < 8; ++nt) {
    const int ncol = (nt << 4) + mA;
    // Preload all 4 B fragments first: one long load clause, graduated waits.
    ABfrag bf[4];
    #pragma unroll
    for (int ks = 0; ks < 4; ++ks)
      #pragma unroll
      for (int j = 0; j < 8; ++j) {
        int kk = ks * 32 + a_koff(j, hf);        // B-layout: same K order as A, pair in wT row
        bf[ks].w[j] = *(const unsigned int*)(wTm + (ncol << 7) + kk);
      }
    v8f c = {};
    #pragma unroll
    for (int ks = 0; ks < 4; ++ks)
      c = __builtin_amdgcn_wmma_f32_16x16x32_bf16(false, a[ks].v, false, bf[ks].v,
                                                  (short)0, c, false, false);
    const float bval = bias[ncol];
    if (mat == 2) {
      // V stored transposed: Vt[(b*128 + h)*512 + key]; C rows are contiguous keys -> u32 stores
      unsigned int* dst = (unsigned int*)(Vt + (((size_t)(b * DIM + ncol)) << 9) + key0 + 8 * hf);
      #pragma unroll
      for (int r2 = 0; r2 < 4; ++r2)
        dst[r2] = pack2bf(c[2 * r2] + bval, c[2 * r2 + 1] + bval);
    } else {
      unsigned short* out = (mat == 0) ? Qb : Kb;
      #pragma unroll
      for (int r = 0; r < 8; ++r) {
        int row = m0 + r + 8 * hf;
        out[((size_t)row << 7) + ncol] = f2bf((c[r] + bval) * scale);
      }
    }
  }
}

// ---------------- kernel 2: flash attention ----------------
// block = 128 threads (4 waves); each wave owns 16 query rows; streams 32-key tiles.
__global__ __launch_bounds__(128) void attn_kernel(
    const unsigned short* __restrict__ Qb, const unsigned short* __restrict__ Kb,
    const unsigned short* __restrict__ Vt, const int* __restrict__ valid_lens,
    float* __restrict__ out) {
  __shared__ unsigned short pscr[4 * 16 * 32];   // per-wave P transpose scratch (1 KB/wave)
  const int lane = threadIdx.x & 31;
  const int wid  = threadIdx.x >> 5;
  const int b    = blockIdx.x >> 3;
  const int r0b  = ((blockIdx.x & 7) << 6) + (wid << 4);  // query-row tile base in batch
  const int mA = lane & 15, hf = lane >> 4;
  const int vl = valid_lens[b];

  // Q A-fragments for all 4 h-steps (held in registers the whole loop)
  ABfrag qf[4];
  const unsigned short* qrow = Qb + (((size_t)(b * SEQ + r0b + mA)) << 7);
  #pragma unroll
  for (int ks = 0; ks < 4; ++ks)
    #pragma unroll
    for (int j = 0; j < 8; ++j) {
      int h0 = ks * 32 + a_koff(j, hf);
      qf[ks].w[j] = *(const unsigned int*)(qrow + h0);
    }

  float m_run[8], l_run[8];
  v8f acc[8];
  const v8f vzero = {};
  #pragma unroll
  for (int r = 0; r < 8; ++r) { m_run[r] = -3.0e38f; l_run[r] = 0.f; }
  #pragma unroll
  for (int ct = 0; ct < 8; ++ct) acc[ct] = vzero;

  unsigned short* myp = pscr + wid * 512;
  const unsigned short* kbase_p = Kb + (((size_t)(b * SEQ + mA)) << 7);
  const unsigned short* vbase_p = Vt + (((size_t)(b * DIM + mA)) << 9);

  for (int kt = 0; kt < 16; ++kt) {
    const int kb = kt << 5;
    // Prefetch next tile's K/V lines into cache (global_prefetch_b8).
    if (kt < 15) {
      __builtin_prefetch(kbase_p + (((size_t)(kb + 32)) << 7), 0, 0);
      __builtin_prefetch(vbase_p + kb + 32, 0, 0);
    }

    // ---- S = Q @ K^T for 32 keys: preload all 8 B fragments, then WMMA chain ----
    v8f s0 = vzero, s1 = vzero;
    {
      const unsigned short* krow0 = kbase_p + ((size_t)kb << 7);
      const unsigned short* krow1 = krow0 + (16 << 7);
      ABfrag kf0[4], kf1[4];
      #pragma unroll
      for (int ks = 0; ks < 4; ++ks)
        #pragma unroll
        for (int j = 0; j < 8; ++j) {
          int h = ks * 32 + a_koff(j, hf);   // (h,h+1) pair in K row, same K order as A
          kf0[ks].w[j] = *(const unsigned int*)(krow0 + h);
          kf1[ks].w[j] = *(const unsigned int*)(krow1 + h);
        }
      #pragma unroll
      for (int ks = 0; ks < 4; ++ks) {
        s0 = __builtin_amdgcn_wmma_f32_16x16x32_bf16(false, qf[ks].v, false, kf0[ks].v,
                                                     (short)0, s0, false, false);
        s1 = __builtin_amdgcn_wmma_f32_16x16x32_bf16(false, qf[ks].v, false, kf1[ks].v,
                                                     (short)0, s1, false, false);
      }
    }
    // ---- column mask (cols >= valid_len -> NEG) ----
    if (kb + mA >= vl) {
      #pragma unroll
      for (int r = 0; r < 8; ++r) s0[r] = NEGV;
    }
    if (kb + 16 + mA >= vl) {
      #pragma unroll
      for (int r = 0; r < 8; ++r) s1[r] = NEGV;
    }

    // ---- online softmax (row stats live in the 16-lane C-layout groups) ----
    float p0[8], p1[8], alpha[8];
    #pragma unroll
    for (int r = 0; r < 8; ++r) {
      float t = fmaxf(s0[r], s1[r]);
      t = fmaxf(t, __shfl_xor(t, 1, 32));
      t = fmaxf(t, __shfl_xor(t, 2, 32));
      t = fmaxf(t, __shfl_xor(t, 4, 32));
      t = fmaxf(t, __shfl_xor(t, 8, 32));
      const float mn = fmaxf(m_run[r], t);
      alpha[r] = __expf(m_run[r] - mn);
      m_run[r] = mn;
      p0[r] = __expf(s0[r] - mn);
      p1[r] = __expf(s1[r] - mn);
      float s = p0[r] + p1[r];
      s += __shfl_xor(s, 1, 32);
      s += __shfl_xor(s, 2, 32);
      s += __shfl_xor(s, 4, 32);
      s += __shfl_xor(s, 8, 32);
      l_run[r] = l_run[r] * alpha[r] + s;
    }
    #pragma unroll
    for (int ct = 0; ct < 8; ++ct)
      #pragma unroll
      for (int r = 0; r < 8; ++r) acc[ct][r] *= alpha[r];

    // ---- P: C-layout -> A-layout via wave-local LDS scratch ----
    #pragma unroll
    for (int r = 0; r < 8; ++r) {
      int row = r + 8 * hf;
      myp[row * 32 + mA]      = f2bf(p0[r]);
      myp[row * 32 + mA + 16] = f2bf(p1[r]);
    }
    asm volatile("s_wait_dscnt 0x0" ::: "memory");   // wave-local: DS counter wait only
    ABfrag pf;
    #pragma unroll
    for (int j = 0; j < 8; ++j) {
      int k0 = a_koff(j, hf);
      pf.w[j] = *(const unsigned int*)(myp + mA * 32 + k0);
    }

    // ---- O += P @ V: preload all 8 V fragments, then WMMA chain ----
    ABfrag vf[8];
    #pragma unroll
    for (int ct = 0; ct < 8; ++ct) {
      const unsigned short* vrow = vbase_p + (((size_t)(ct << 4)) << 9) + kb;
      #pragma unroll
      for (int j = 0; j < 8; ++j) {
        int kk = a_koff(j, hf);
        vf[ct].w[j] = *(const unsigned int*)(vrow + kk);
      }
    }
    #pragma unroll
    for (int ct = 0; ct < 8; ++ct)
      acc[ct] = __builtin_amdgcn_wmma_f32_16x16x32_bf16(false, pf.v, false, vf[ct].v,
                                                        (short)0, acc[ct], false, false);
  }

  // ---- epilogue: normalize; rows >= valid_len -> 0 (matches reference mask) ----
  #pragma unroll
  for (int r = 0; r < 8; ++r) {
    const int rowb = r0b + r + 8 * hf;
    const float inv = (rowb < vl) ? (1.0f / l_run[r]) : 0.0f;
    #pragma unroll
    for (int ct = 0; ct < 8; ++ct)
      out[(((size_t)(b * SEQ + rowb)) << 7) + (ct << 4) + mA] = acc[ct][r] * inv;
  }
}

extern "C" void kernel_launch(void* const* d_in, const int* in_sizes, int n_in,
                              void* d_out, int out_size, void* d_ws, size_t ws_size,
                              hipStream_t stream) {
  (void)in_sizes; (void)n_in; (void)out_size; (void)ws_size;
  const float* x  = (const float*)d_in[0];
  const float* wq = (const float*)d_in[1];
  const float* bq = (const float*)d_in[2];
  const float* wk = (const float*)d_in[3];
  const float* bk = (const float*)d_in[4];
  const float* wv = (const float*)d_in[5];
  const float* bv = (const float*)d_in[6];
  const int*   vl = (const int*)d_in[7];
  float* out = (float*)d_out;

  unsigned short* ws = (unsigned short*)d_ws;
  unsigned short* wT = ws;                                   // 3*128*128 bf16
  unsigned short* Qb = wT + (size_t)3 * 128 * 128;           // 256*512*128 bf16
  unsigned short* Kb = Qb + (size_t)BATCH * SEQ * DIM;
  unsigned short* Vt = Kb + (size_t)BATCH * SEQ * DIM;       // transposed V

  prep_weights<<<192, 256, 0, stream>>>(wq, wk, wv, wT);
  qkv_proj<<<3072, 256, 0, stream>>>(x, wT, bq, bk, bv, Qb, Kb, Vt);
  attn_kernel<<<BATCH * 8, 128, 0, stream>>>(Qb, Kb, Vt, vl, out);
}
